// InvariantPolynomial_54846732369989
// MI455X (gfx1250) — matrix operand order repaired and checked by reading, not compiled, not run on hardware
//
#include <hip/hip_runtime.h>
#include <hip/hip_bf16.h>
#include <math.h>

#define N_NODES 50000
#define N_EDGES 600000
#define N_MOL   512
#define N_AF    30
#define C_TOT   104   // 64 + 24 + 16
#define F_TOT   216   // 64 + 72 + 80

typedef __attribute__((ext_vector_type(16))) _Float16 v16h;
typedef __attribute__((ext_vector_type(8)))  float    v8f;

// ---------------------------------------------------------------------------
// Helper: concatenated W1 lookup  W1 = [W1_0 | W1_1 | W1_2]  (30 x 104)
// ---------------------------------------------------------------------------
__device__ __forceinline__ float w1_lookup(const float* __restrict__ W1_0,
                                           const float* __restrict__ W1_1,
                                           const float* __restrict__ W1_2,
                                           int k, int c) {
    if (c < 64) return W1_0[k * 64 + c];
    if (c < 88) return W1_1[k * 24 + (c - 64)];
    return W1_2[k * 16 + (c - 88)];
}

// ---------------------------------------------------------------------------
// K1: node_a = (x @ W1) * (1/sqrt(30))  via v_wmma_f32_16x16x32_f16
// One wave per 16-node tile; loops over 7 column tiles of 16 (covers 104).
// ---------------------------------------------------------------------------
__global__ void k_gemm_xw1(const float* __restrict__ x,
                           const float* __restrict__ W1_0,
                           const float* __restrict__ W1_1,
                           const float* __restrict__ W1_2,
                           float* __restrict__ node_a) {
    const int lane = threadIdx.x & 31;
    const int tile = blockIdx.x * (blockDim.x >> 5) + (threadIdx.x >> 5);
    if (tile >= N_NODES / 16) return;

    const int hi  = lane >> 4;   // half-wave select
    const int lm  = lane & 15;
    const int row = tile * 16 + lm;   // M for A-matrix = lane&15

    // --- A: 16x32 tile of x (K padded 30 -> 32 with zeros) -----------------
    // ISA 16-bit A 16x32 layout: halves 0..7 -> K = hi*8 + i,
    //                            halves 8..15 -> K = 16 + hi*8 + (i-8)
    v16h A;
#pragma unroll
    for (int i = 0; i < 8; ++i) {
        const int k = hi * 8 + i;                    // always < 16 < 30
        A[i] = (_Float16)x[row * N_AF + k];
    }
#pragma unroll
    for (int i = 8; i < 16; ++i) {
        const int k = 16 + hi * 8 + (i - 8);         // 16..31
        A[i] = (k < N_AF) ? (_Float16)x[row * N_AF + k] : (_Float16)0.f;
    }

    const float scale = 0.18257418583505536f;        // 1/sqrt(30)

    for (int ct = 0; ct < 7; ++ct) {
        const int cb  = ct * 16;
        const int col = cb + lm;                     // N for B/C/D = lane&15

        // --- B: 32x16 tile of W1 (K x N); halves i -> K = hi*8 + i (i<8), 16 + hi*8 + (i-8) (i>=8) --------
        v16h B;
#pragma unroll
        for (int i = 0; i < 16; ++i) {
            const int k = (i < 8) ? (hi * 8 + i) : (16 + hi * 8 + (i - 8));
            B[i] = (k < N_AF && col < C_TOT)
                       ? (_Float16)w1_lookup(W1_0, W1_1, W1_2, k, col)
                       : (_Float16)0.f;
        }

        v8f C = {};
        C = __builtin_amdgcn_wmma_f32_16x16x32_f16(
                /*neg_a=*/false, A, /*neg_b=*/false, B,
                /*c_mod=*/(short)0, C, /*reuse_a=*/false, /*reuse_b=*/false);

        if (col < C_TOT) {
#pragma unroll
            for (int r = 0; r < 8; ++r) {
                const int m = r + hi * 8;            // D layout: M = r + 8*hi
                node_a[(size_t)(tile * 16 + m) * C_TOT + col] = C[r] * scale;
            }
        }
    }
}

// ---------------------------------------------------------------------------
// ea[0..8] = edge_attr[e,0] * sh(edge_vec)[0..8]   (only l<=2 terms are used)
// ---------------------------------------------------------------------------
__device__ __forceinline__ void compute_ea(const float* __restrict__ pos,
                                           const float* __restrict__ edge_attr,
                                           int e, int s, int d, float ea[9]) {
    const float vx = pos[s * 3 + 0] - pos[d * 3 + 0];
    const float vy = pos[s * 3 + 1] - pos[d * 3 + 1];
    const float vz = pos[s * 3 + 2] - pos[d * 3 + 2];
    const float r2 = vx * vx + vy * vy + vz * vz;
    const float s3  = 1.7320508075688772f;   // sqrt(3)
    const float s5  = 2.23606797749979f;     // sqrt(5)
    const float s15 = 3.872983346207417f;    // sqrt(15)
    const float w = edge_attr[(size_t)e * 10];
    ea[0] = w;
    ea[1] = w * s3 * vx;
    ea[2] = w * s3 * vy;
    ea[3] = w * s3 * vz;
    ea[4] = w * s15 * vx * vy;
    ea[5] = w * s15 * vy * vz;
    ea[6] = w * 0.5f * s5 * (3.f * vz * vz - r2);
    ea[7] = w * s15 * vx * vz;
    ea[8] = w * 0.5f * s15 * (vx * vx - vy * vy);
}

// ---------------------------------------------------------------------------
// K2: wave-per-edge. e_feat (216) scattered into nsum[dst] with coalesced
// float atomics (lanes cover contiguous components of one row). deg += 1.
// ---------------------------------------------------------------------------
__global__ void k_edge_scatter(const float* __restrict__ pos,
                               const float* __restrict__ edge_attr,
                               const int* __restrict__ edge_src,
                               const int* __restrict__ edge_dst,
                               const float* __restrict__ node_a,
                               float* __restrict__ nsum,
                               float* __restrict__ deg) {
    const int wid  = (blockIdx.x * blockDim.x + threadIdx.x) >> 5;
    const int lane = threadIdx.x & 31;
    if (wid >= N_EDGES) return;
    const int s = edge_src[wid];
    const int d = edge_dst[wid];

    float ea[9];
    compute_ea(pos, edge_attr, wid, s, d, ea);

    const float* arow = node_a + (size_t)s * C_TOT;
    float*       nrow = nsum   + (size_t)d * F_TOT;

    for (int j = lane; j < F_TOT; j += 32) {
        float av, ev;
        if (j < 64)        { av = arow[j];                 ev = ea[0]; }
        else if (j < 136)  { const int r = j - 64;
                             av = arow[64 + r / 3];        ev = ea[1 + r % 3]; }
        else               { const int r = j - 136;
                             av = arow[88 + r / 5];        ev = ea[4 + r % 5]; }
        atomicAdd(&nrow[j], av * ev);
    }
    if (lane == 0) atomicAdd(&deg[d], 1.0f);
}

// ---------------------------------------------------------------------------
// K3: wave-per-edge. g = nsum[src]*isd(src); contract with ea and W2;
// wave reduction; one atomic per edge into node_out[dst].
// ---------------------------------------------------------------------------
__global__ void k_edge_reduce(const float* __restrict__ pos,
                              const float* __restrict__ edge_attr,
                              const int* __restrict__ edge_src,
                              const int* __restrict__ edge_dst,
                              const float* __restrict__ nsum,
                              const float* __restrict__ deg,
                              const float* __restrict__ W2_0,
                              const float* __restrict__ W2_1,
                              const float* __restrict__ W2_2,
                              float* __restrict__ node_out) {
    const int wid  = (blockIdx.x * blockDim.x + threadIdx.x) >> 5;
    const int lane = threadIdx.x & 31;
    if (wid >= N_EDGES) return;
    const int s = edge_src[wid];
    const int d = edge_dst[wid];

    float ea[9];
    compute_ea(pos, edge_attr, wid, s, d, ea);

    const float dg  = deg[s];
    const float isd = (dg > 0.f) ? rsqrtf(fmaxf(dg, 1.f)) : 0.f;
    const float inv_s3 = 0.5773502691896258f;   // 1/sqrt(3)
    const float inv_s5 = 0.4472135954999579f;   // 1/sqrt(5)

    const float* nrow = nsum + (size_t)s * F_TOT;
    float part = 0.f;
    for (int j = lane; j < F_TOT; j += 32) {
        const float g = nrow[j] * isd;
        if (j < 64)        { part += g * ea[0]         * W2_0[j]; }
        else if (j < 136)  { const int r = j - 64;
                             part += g * ea[1 + r % 3] * W2_1[r / 3] * inv_s3; }
        else               { const int r = j - 136;
                             part += g * ea[4 + r % 5] * W2_2[r / 5] * inv_s5; }
    }
#pragma unroll
    for (int off = 16; off > 0; off >>= 1)
        part += __shfl_xor(part, off, 32);

    if (lane == 0)
        atomicAdd(&node_out[d], part * 0.09805806756909202f);  // 1/sqrt(104)
}

// ---------------------------------------------------------------------------
// K4: per-node -> per-molecule segment sums
// ---------------------------------------------------------------------------
__global__ void k_node_mol(const float* __restrict__ node_out,
                           const float* __restrict__ deg,
                           const int* __restrict__ batch,
                           float* __restrict__ mol_acc,
                           float* __restrict__ mol_cnt) {
    const int n = blockIdx.x * blockDim.x + threadIdx.x;
    if (n >= N_NODES) return;
    const float dg  = deg[n];
    const float isd = (dg > 0.f) ? rsqrtf(fmaxf(dg, 1.f)) : 0.f;
    const int   b   = batch[n];
    atomicAdd(&mol_acc[b], node_out[n] * isd);
    atomicAdd(&mol_cnt[b], 1.0f);
}

// ---------------------------------------------------------------------------
// K5: out[m] = mol_acc[m] * rsqrt(atoms) (0 if empty)
// ---------------------------------------------------------------------------
__global__ void k_finalize(const float* __restrict__ mol_acc,
                           const float* __restrict__ mol_cnt,
                           float* __restrict__ out) {
    const int m = blockIdx.x * blockDim.x + threadIdx.x;
    if (m >= N_MOL) return;
    const float c = mol_cnt[m];
    out[m] = mol_acc[m] * ((c > 0.f) ? rsqrtf(fmaxf(c, 1.f)) : 0.f);
}

// ---------------------------------------------------------------------------
extern "C" void kernel_launch(void* const* d_in, const int* in_sizes, int n_in,
                              void* d_out, int out_size, void* d_ws, size_t ws_size,
                              hipStream_t stream) {
    (void)in_sizes; (void)n_in; (void)out_size; (void)ws_size;

    const float* positions = (const float*)d_in[0];
    const float* x         = (const float*)d_in[1];
    const float* edge_attr = (const float*)d_in[2];
    const float* W1_0      = (const float*)d_in[3];
    const float* W1_1      = (const float*)d_in[4];
    const float* W1_2      = (const float*)d_in[5];
    const float* W2_0      = (const float*)d_in[6];
    const float* W2_1      = (const float*)d_in[7];
    const float* W2_2      = (const float*)d_in[8];
    const int*   edge_src  = (const int*)d_in[9];
    const int*   edge_dst  = (const int*)d_in[10];
    const int*   batch     = (const int*)d_in[11];
    float* out = (float*)d_out;

    // Workspace layout (floats): ~64.4 MB total
    float* ws       = (float*)d_ws;
    float* node_a   = ws;                                      // 50000*104
    float* nsum     = node_a + (size_t)N_NODES * C_TOT;        // 50000*216
    float* deg      = nsum   + (size_t)N_NODES * F_TOT;        // 50000
    float* node_out = deg + N_NODES;                           // 50000
    float* mol_acc  = node_out + N_NODES;                      // 512
    float* mol_cnt  = mol_acc + N_MOL;                         // 512

    // Zero all accumulation buffers (node_a is fully overwritten by K1).
    const size_t zero_bytes =
        ((size_t)N_NODES * F_TOT + 2 * (size_t)N_NODES + 2 * (size_t)N_MOL) * sizeof(float);
    hipMemsetAsync(nsum, 0, zero_bytes, stream);

    // K1: WMMA GEMM. 3125 row tiles, 4 waves / block of 128.
    k_gemm_xw1<<<(N_NODES / 16 + 3) / 4, 128, 0, stream>>>(x, W1_0, W1_1, W1_2, node_a);

    // K2/K3: wave-per-edge, 8 waves per 256-thread block.
    k_edge_scatter<<<N_EDGES / 8, 256, 0, stream>>>(
        positions, edge_attr, edge_src, edge_dst, node_a, nsum, deg);
    k_edge_reduce<<<N_EDGES / 8, 256, 0, stream>>>(
        positions, edge_attr, edge_src, edge_dst, nsum, deg, W2_0, W2_1, W2_2, node_out);

    k_node_mol<<<(N_NODES + 255) / 256, 256, 0, stream>>>(node_out, deg, batch, mol_acc, mol_cnt);
    k_finalize<<<(N_MOL + 255) / 256, 256, 0, stream>>>(mol_acc, mol_cnt, out);
}
